// SupervisedModel_2027224563913
// MI455X (gfx1250) — matrix operand order repaired and checked by reading, not compiled, not run on hardware
//
#include <hip/hip_runtime.h>

typedef __bf16 bf16_t;
typedef __attribute__((ext_vector_type(16))) __bf16 v16bf;
typedef __attribute__((ext_vector_type(8)))  __bf16 v8bf;
typedef __attribute__((ext_vector_type(8)))  float  v8f;

#define SEQ   200
#define BATCH 32768

#if __has_builtin(__builtin_amdgcn_tanhf)
__device__ __forceinline__ float fast_tanh(float v) { return __builtin_amdgcn_tanhf(v); }
#elif __has_builtin(__builtin_amdgcn_tanh_f32)
__device__ __forceinline__ float fast_tanh(float v) { return __builtin_amdgcn_tanh_f32(v); }
#else
__device__ __forceinline__ float fast_tanh(float v) { return tanhf(v); }
#endif

__device__ __forceinline__ float sigf(float v) { return 0.5f + 0.5f * fast_tanh(0.5f * v); }

// ---------------------------------------------------------------------------
// f32 -> bf16 weight convert with zero-padded destination stride.
// ---------------------------------------------------------------------------
__global__ void cvt_pad_kernel(const float* __restrict__ src, bf16_t* __restrict__ dst,
                               int rows, int cols, int dstride) {
    int idx = blockIdx.x * blockDim.x + threadIdx.x;
    int total = rows * dstride;
    if (idx >= total) return;
    int r = idx / dstride;
    int c = idx - r * dstride;
    dst[idx] = (c < cols) ? (bf16_t)src[r * cols + c] : (bf16_t)0.0f;
}

// ---------------------------------------------------------------------------
// 2-layer LSTM (H=4), one thread per sample for one channel.
// Writes layer-1 output h1[t] (bf16, packed 8B store) to hseq[b][t*4+k]
// (stride 800) and x (bf16) to xbf[b][ch*200+t] (stride 416, pads on ch==1).
// ---------------------------------------------------------------------------
__global__ __launch_bounds__(256) void lstm2_kernel(
    const float* __restrict__ x,
    const float* __restrict__ Wih0, const float* __restrict__ Whh0,
    const float* __restrict__ bih0, const float* __restrict__ bhh0,
    const float* __restrict__ Wih1, const float* __restrict__ Whh1,
    const float* __restrict__ bih1, const float* __restrict__ bhh1,
    bf16_t* __restrict__ hseq, bf16_t* __restrict__ xbf, int ch) {

    __shared__ float s_wx0[16], s_b0[16], s_b1[16];
    __shared__ float s_whh0[16][4], s_wih1[16][4], s_whh1[16][4];
    int tid = threadIdx.x;
    if (tid < 16) {
        s_wx0[tid] = Wih0[tid];
        s_b0[tid]  = bih0[tid] + bhh0[tid];
        s_b1[tid]  = bih1[tid] + bhh1[tid];
        for (int k = 0; k < 4; ++k) {
            s_whh0[tid][k] = Whh0[tid * 4 + k];
            s_wih1[tid][k] = Wih1[tid * 4 + k];
            s_whh1[tid][k] = Whh1[tid * 4 + k];
        }
    }
    __syncthreads();

    int b = blockIdx.x * blockDim.x + tid;
    if (b >= BATCH) return;

    const float* xp = x + (size_t)(b * 2 + ch) * SEQ;
    bf16_t* hp = hseq + (size_t)b * 800;
    bf16_t* xo = xbf  + (size_t)b * 416 + ch * SEQ;

    float h0[4] = {0.f,0.f,0.f,0.f}, c0[4] = {0.f,0.f,0.f,0.f};
    float h1[4] = {0.f,0.f,0.f,0.f}, c1[4] = {0.f,0.f,0.f,0.f};

    union HPack { bf16_t h[4]; uint2 u; };

    for (int t = 0; t < SEQ; ++t) {
        float xt = xp[t];
        xo[t] = (bf16_t)xt;

        float g[16];
        #pragma unroll
        for (int j = 0; j < 16; ++j)
            g[j] = s_b0[j] + s_wx0[j] * xt
                 + s_whh0[j][0]*h0[0] + s_whh0[j][1]*h0[1]
                 + s_whh0[j][2]*h0[2] + s_whh0[j][3]*h0[3];
        #pragma unroll
        for (int k = 0; k < 4; ++k) {
            float ig = sigf(g[k]), fg = sigf(g[4+k]);
            float gg = fast_tanh(g[8+k]), og = sigf(g[12+k]);
            c0[k] = fg * c0[k] + ig * gg;
            h0[k] = og * fast_tanh(c0[k]);
        }
        #pragma unroll
        for (int j = 0; j < 16; ++j)
            g[j] = s_b1[j]
                 + s_wih1[j][0]*h0[0] + s_wih1[j][1]*h0[1]
                 + s_wih1[j][2]*h0[2] + s_wih1[j][3]*h0[3]
                 + s_whh1[j][0]*h1[0] + s_whh1[j][1]*h1[1]
                 + s_whh1[j][2]*h1[2] + s_whh1[j][3]*h1[3];
        #pragma unroll
        for (int k = 0; k < 4; ++k) {
            float ig = sigf(g[k]), fg = sigf(g[4+k]);
            float gg = fast_tanh(g[8+k]), og = sigf(g[12+k]);
            c1[k] = fg * c1[k] + ig * gg;
            h1[k] = og * fast_tanh(c1[k]);
        }
        HPack pk;
        #pragma unroll
        for (int k = 0; k < 4; ++k) pk.h[k] = (bf16_t)h1[k];
        *(uint2*)(hp + t * 4) = pk.u;
    }
    if (ch == 1) {
        bf16_t* padp = xbf + (size_t)b * 416 + 400;
        #pragma unroll
        for (int k = 0; k < 16; ++k) padp[k] = (bf16_t)0.0f;
    }
}

// ---------------------------------------------------------------------------
// WMMA bf16 GEMM:  out = act(A[M,K] * Bw[N,K]^T + bias)
// Compile-time NT (N tiles of 16), MODE, PAD16 -> branch-free hot loop.
// Each wave computes 2 M tiles of 16 rows (32 rows) x NT*16 cols.
// MODE 0: tanh -> bf16 out; MODE 1: plain -> f32 out (col < Nvalid);
// MODE 2: tanh -> f32 out (col < Nvalid).
// PAD16: zero bf16 out columns [Nvalid, Nvalid+16).
// ---------------------------------------------------------------------------
template <int NT, int MODE, int PAD16>
__global__ __launch_bounds__(128) void gemm_wmma_kernel(
    const bf16_t* __restrict__ A, int lda,
    const bf16_t* __restrict__ Bw, int ldb,
    const float* __restrict__ bias,
    void* __restrict__ out, int ldo,
    int K, int Nvalid) {

    const int lane   = threadIdx.x & 31;
    const int wave   = threadIdx.x >> 5;
    const int m0     = (blockIdx.x * 4 + wave) * 32;   // 2 M tiles per wave
    const int hi     = (lane >= 16) ? 1 : 0;
    const int nlane  = lane & 15;
    const int a_koff = hi ? 8 : 0;    // A frag: low half K0..7/K16..23, hi half K8..15/K24..31
    const int b_koff = hi ? 8 : 0;    // B frag: low half K0..7/K16..23, hi half K8..15/K24..31

    union U { v16bf v; v8bf h[2]; };

    v8f acc[2][NT];
    #pragma unroll
    for (int mt = 0; mt < 2; ++mt)
        #pragma unroll
        for (int nt = 0; nt < NT; ++nt) acc[mt][nt] = (v8f){};

    const bf16_t* aptr0 = A + (size_t)(m0 + nlane) * lda + a_koff;
    const bf16_t* aptr1 = aptr0 + (size_t)16 * lda;
    const bf16_t* bptr  = Bw + (size_t)nlane * ldb + b_koff;

    for (int kc = 0; kc < K; kc += 32) {
        U a[2], bb[NT];
        a[0].h[0] = *(const v8bf*)(aptr0 + kc);
        a[0].h[1] = *(const v8bf*)(aptr0 + kc + 16);
        a[1].h[0] = *(const v8bf*)(aptr1 + kc);
        a[1].h[1] = *(const v8bf*)(aptr1 + kc + 16);
        #pragma unroll
        for (int nt = 0; nt < NT; ++nt) {
            const bf16_t* bp = bptr + (size_t)(nt * 16) * ldb + kc;
            bb[nt].h[0] = *(const v8bf*)(bp);
            bb[nt].h[1] = *(const v8bf*)(bp + 16);
        }
        #pragma unroll
        for (int nt = 0; nt < NT; ++nt) {
            acc[0][nt] = __builtin_amdgcn_wmma_f32_16x16x32_bf16(
                false, a[0].v, false, bb[nt].v, (short)0, acc[0][nt], false, false);
            acc[1][nt] = __builtin_amdgcn_wmma_f32_16x16x32_bf16(
                false, a[1].v, false, bb[nt].v, (short)0, acc[1][nt], false, false);
        }
    }

    #pragma unroll
    for (int mt = 0; mt < 2; ++mt) {
        const int rbase = m0 + mt * 16 + (hi ? 8 : 0);
        #pragma unroll
        for (int nt = 0; nt < NT; ++nt) {
            int col = nt * 16 + nlane;
            float bv = (col < Nvalid) ? bias[col] : 0.0f;
            if (MODE == 0) {
                bf16_t* op = (bf16_t*)out;
                #pragma unroll
                for (int j = 0; j < 8; ++j)
                    op[(size_t)(rbase + j) * ldo + col] =
                        (bf16_t)fast_tanh(acc[mt][nt][j] + bv);
            } else if (MODE == 1) {
                float* op = (float*)out;
                if (col < Nvalid) {
                    #pragma unroll
                    for (int j = 0; j < 8; ++j)
                        op[(size_t)(rbase + j) * ldo + col] = acc[mt][nt][j] + bv;
                }
            } else {
                float* op = (float*)out;
                if (col < Nvalid) {
                    #pragma unroll
                    for (int j = 0; j < 8; ++j)
                        op[(size_t)(rbase + j) * ldo + col] =
                            fast_tanh(acc[mt][nt][j] + bv);
                }
            }
        }
        if (PAD16) {
            bf16_t* op = (bf16_t*)out;
            int col = Nvalid + nlane;
            #pragma unroll
            for (int j = 0; j < 8; ++j)
                op[(size_t)(rbase + j) * ldo + col] = (bf16_t)0.0f;
        }
    }
}

// ---------------------------------------------------------------------------
// Head: out[b] = sigmoid(b3 + z2[b]·W3[0:40] + other[b]·W3[40:80])
// (z2 already has tanh applied by MODE-2 GEMM store.)
// ---------------------------------------------------------------------------
__global__ __launch_bounds__(256) void head_kernel(
    const float* __restrict__ z2, const float* __restrict__ other,
    const float* __restrict__ W3, const float* __restrict__ b3,
    float* __restrict__ out) {
    int b = blockIdx.x * blockDim.x + threadIdx.x;
    if (b >= BATCH) return;
    const float* zp = z2 + (size_t)b * 40;
    const float* op = other + (size_t)b * 40;
    float s = b3[0];
    #pragma unroll 8
    for (int j = 0; j < 40; ++j) s += zp[j] * W3[j];
    #pragma unroll 8
    for (int j = 0; j < 40; ++j) s += op[j] * W3[40 + j];
    out[b] = sigf(s);
}

// ---------------------------------------------------------------------------
extern "C" void kernel_launch(void* const* d_in, const int* in_sizes, int n_in,
                              void* d_out, int out_size, void* d_ws, size_t ws_size,
                              hipStream_t stream) {
    const float* x = (const float*)d_in[0];
    const float* r0w[8]; for (int i = 0; i < 8; ++i) r0w[i] = (const float*)d_in[1 + i];
    const float* r1w[8]; for (int i = 0; i < 8; ++i) r1w[i] = (const float*)d_in[9 + i];
    const float* Wp = (const float*)d_in[17]; const float* bp = (const float*)d_in[18];
    const float* WH = (const float*)d_in[19]; const float* bH = (const float*)d_in[20];
    const float* WL = (const float*)d_in[21]; const float* bL = (const float*)d_in[22];
    const float* W1 = (const float*)d_in[23]; const float* b1 = (const float*)d_in[24];
    const float* W2 = (const float*)d_in[25]; const float* b2 = (const float*)d_in[26];
    const float* W3 = (const float*)d_in[27]; const float* b3 = (const float*)d_in[28];
    float* out = (float*)d_out;

    // Workspace carve-up (all chunks 16B-aligned by construction).
    char* w = (char*)d_ws;
    bf16_t* hseq = (bf16_t*)w; w += (size_t)BATCH * 800 * 2;   // reused H then L
    bf16_t* xbf  = (bf16_t*)w; w += (size_t)BATCH * 416 * 2;   // padded x_flat
    bf16_t* zbf  = (bf16_t*)w; w += (size_t)BATCH * 256 * 2;   // [Hf | Lf]
    bf16_t* z1bf = (bf16_t*)w; w += (size_t)BATCH * 96  * 2;   // tanh(z@W1)+pad
    float*  z2f  = (float*)w;  w += (size_t)BATCH * 40  * 4;   // tanh(z1@W2+b2)
    float*  othf = (float*)w;  w += (size_t)BATCH * 40  * 4;   // x_flat@Wp+bp
    bf16_t* whbf = (bf16_t*)w; w += (size_t)128 * 800 * 2;
    bf16_t* wlbf = (bf16_t*)w; w += (size_t)128 * 800 * 2;
    bf16_t* w1bf = (bf16_t*)w; w += (size_t)80  * 256 * 2;
    bf16_t* w2bf = (bf16_t*)w; w += (size_t)40  * 96  * 2;
    bf16_t* wpbf = (bf16_t*)w; w += (size_t)40  * 416 * 2;

    // 1) bf16 weight conversion (zero-padded K strides for W2 / Wp).
    cvt_pad_kernel<<<(128*800 + 255) / 256, 256, 0, stream>>>(WH, whbf, 128, 800, 800);
    cvt_pad_kernel<<<(128*800 + 255) / 256, 256, 0, stream>>>(WL, wlbf, 128, 800, 800);
    cvt_pad_kernel<<<(80*256  + 255) / 256, 256, 0, stream>>>(W1, w1bf, 80, 256, 256);
    cvt_pad_kernel<<<(40*96   + 255) / 256, 256, 0, stream>>>(W2, w2bf, 40, 80, 96);
    cvt_pad_kernel<<<(40*416  + 255) / 256, 256, 0, stream>>>(Wp, wpbf, 40, 400, 416);

    const int lstm_blocks = BATCH / 256;
    const int gemm_blocks = BATCH / 128;   // 4 waves/block, 32 rows/wave

    // 2) H channel: LSTM -> hseq, then Hf = tanh(hseq@WH^T+bH) -> zbf[:,0:128]
    lstm2_kernel<<<lstm_blocks, 256, 0, stream>>>(x, r0w[0], r0w[1], r0w[2], r0w[3],
                                                  r0w[4], r0w[5], r0w[6], r0w[7],
                                                  hseq, xbf, 0);
    gemm_wmma_kernel<8, 0, 0><<<gemm_blocks, 128, 0, stream>>>(
        hseq, 800, whbf, 800, bH, zbf, 256, 800, 128);

    // 3) L channel (reuses hseq): Lf -> zbf[:,128:256]
    lstm2_kernel<<<lstm_blocks, 256, 0, stream>>>(x, r1w[0], r1w[1], r1w[2], r1w[3],
                                                  r1w[4], r1w[5], r1w[6], r1w[7],
                                                  hseq, xbf, 1);
    gemm_wmma_kernel<8, 0, 0><<<gemm_blocks, 128, 0, stream>>>(
        hseq, 800, wlbf, 800, bL, zbf + 128, 256, 800, 128);

    // 4) other = x_flat @ Wp^T + bp  (K padded 400->416), plain f32
    gemm_wmma_kernel<3, 1, 0><<<gemm_blocks, 128, 0, stream>>>(
        xbf, 416, wpbf, 416, bp, othf, 40, 416, 40);

    // 5) z1 = tanh(z @ W1^T + b1) -> bf16, stride 96 with zeroed pad cols 80..95
    gemm_wmma_kernel<5, 0, 1><<<gemm_blocks, 128, 0, stream>>>(
        zbf, 256, w1bf, 256, b1, z1bf, 96, 256, 80);

    // 6) z2 = tanh(z1 @ W2^T + b2) -> f32 (K padded 80->96)
    gemm_wmma_kernel<3, 2, 0><<<gemm_blocks, 128, 0, stream>>>(
        z1bf, 96, w2bf, 96, b2, z2f, 40, 96, 40);

    // 7) head: out = sigmoid(z2·W3[0:40] + other·W3[40:80] + b3)
    head_kernel<<<(BATCH + 255) / 256, 256, 0, stream>>>(z2f, othf, W3, b3, out);
}
